// GCAModel_40707700031609
// MI455X (gfx1250) — matrix operand order repaired and checked by reading, not compiled, not run on hardware
//
#include <hip/hip_runtime.h>
#include <hip/hip_bf16.h>

// ---- model dims ----
#define B_   2
#define S_   2048
#define H_   1024
#define NH_  8
#define HD_  128
#define L_   2
#define V_   32000
#define CS_  128
#define NC_  16
#define TOPK_ 8
#define MROWS (B_*S_)   // 4096

typedef __attribute__((ext_vector_type(8)))  float    v8f;
typedef __attribute__((ext_vector_type(16))) _Float16 v16h;
typedef __attribute__((ext_vector_type(8)))  _Float16 v8h;
typedef __attribute__((ext_vector_type(2)))  _Float16 v2h;

__device__ __forceinline__ v8f zero8(){
  v8f z;
  #pragma unroll
  for(int i=0;i<8;i++) z[i]=0.f;
  return z;
}

// pack two float4 (8 f32) into 8 f16
__device__ __forceinline__ v8h pack8(float4 a, float4 b){
  v8h h;
  h[0]=(_Float16)a.x; h[1]=(_Float16)a.y; h[2]=(_Float16)a.z; h[3]=(_Float16)a.w;
  h[4]=(_Float16)b.x; h[5]=(_Float16)b.y; h[6]=(_Float16)b.z; h[7]=(_Float16)b.w;
  return h;
}

// assemble a 16-element f16 WMMA operand from two aligned 8-element LDS chunks
__device__ __forceinline__ v16h mk_op(const _Float16* lo, const _Float16* hi){
  v8h a = *(const v8h*)lo;
  v8h b = *(const v8h*)hi;
  v16h r;
  #pragma unroll
  for(int i=0;i<8;i++){ r[i] = a[i]; r[i+8] = b[i]; }
  return r;
}

// =====================================================================
// Generic GEMM: out[M,N] = act(A[M,K] @ W[K,N] + bias) (+ res)
// act: 0=none, 1=relu, 2=gelu(exact erf)
// Tiles: 128x128x32, 256 threads (8 waves, 2x4), f16 WMMA, f32 accum.
// =====================================================================
__global__ __launch_bounds__(256) void gemm_bias_act_kernel(
    const float* __restrict__ A, const float* __restrict__ W,
    const float* __restrict__ bias, const float* __restrict__ res,
    float* __restrict__ out, int M, int N, int K, int act)
{
  __shared__ _Float16 As[128*40];  // [m][k], pitch 40 (16B aligned rows)
  __shared__ _Float16 Bs[128*40];  // [n][k], pitch 40

  const int tid   = threadIdx.x;
  const int lane  = tid & 31;
  const int wave  = tid >> 5;
  const int waveM = wave >> 2;     // 0..1
  const int waveN = wave & 3;      // 0..3
  const int bm = blockIdx.y, bn = blockIdx.x;
  const int l15 = lane & 15;
  const int lh  = lane >> 4;

  // A-staging mapping: 2 threads per row, 16 cols each
  const int a_row = tid >> 1;
  const int a_c0  = (tid & 1) * 16;
  const int a_gr  = bm*128 + a_row;
  const bool a_ok = (a_gr < M);
  // B-staging mapping: thread = (k-pair, 8 columns)
  const int b_kp = tid & 15;        // k = 2*b_kp
  const int b_n0 = (tid >> 4) * 8;  // 0..120

  v8f acc[4][2];
  #pragma unroll
  for(int mt=0;mt<4;mt++)
    #pragma unroll
    for(int nt=0;nt<2;nt++) acc[mt][nt] = zero8();

  for(int k0=0;k0<K;k0+=32){
    __syncthreads();
    // stage A tile 128x32 (f32 -> f16), vectorized
    {
      v8h* dst = (v8h*)&As[a_row*40 + a_c0];
      if (a_ok){
        const float4* src = (const float4*)(A + (size_t)a_gr*K + k0 + a_c0);
        float4 f0 = src[0], f1 = src[1], f2 = src[2], f3 = src[3];
        dst[0] = pack8(f0, f1);
        dst[1] = pack8(f2, f3);
      } else {
        v8h z;
        #pragma unroll
        for(int i=0;i<8;i++) z[i] = (_Float16)0.f;
        dst[0] = z; dst[1] = z;
      }
    }
    // stage B tile 32x128 transposed to [n][k]; pack k-pairs -> b32 stores
    {
      const float4* s0 = (const float4*)(W + (size_t)(k0 + 2*b_kp    )*N + bn*128 + b_n0);
      const float4* s1 = (const float4*)(W + (size_t)(k0 + 2*b_kp + 1)*N + bn*128 + b_n0);
      float4 a0 = s0[0], a1 = s0[1];
      float4 c0 = s1[0], c1 = s1[1];
      float w0[8] = {a0.x,a0.y,a0.z,a0.w,a1.x,a1.y,a1.z,a1.w};
      float w1[8] = {c0.x,c0.y,c0.z,c0.w,c1.x,c1.y,c1.z,c1.w};
      #pragma unroll
      for(int j=0;j<8;j++){
        v2h t; t[0] = (_Float16)w0[j]; t[1] = (_Float16)w1[j];
        *(v2h*)&Bs[(b_n0+j)*40 + 2*b_kp] = t;
      }
    }
    if (k0 + 32 < K)
      __builtin_prefetch((const void*)(W + (size_t)(k0+32)*N + bn*128 + b_n0), 0, 1);
    __syncthreads();

    // A operands: row = l15, K set = {lh*8..+7} U {16+lh*8..+7}
    v16h aop[4];
    #pragma unroll
    for(int mt=0;mt<4;mt++){
      const _Float16* p = &As[(waveM*64 + mt*16 + l15)*40 + lh*8];
      aop[mt] = mk_op(p, p+16);
    }
    #pragma unroll
    for(int nt=0;nt<2;nt++){
      const _Float16* p = &Bs[(waveN*32 + nt*16 + l15)*40 + lh*8];
      v16h bop = mk_op(p, p+16);
      #pragma unroll
      for(int mt=0;mt<4;mt++)
        acc[mt][nt] = __builtin_amdgcn_wmma_f32_16x16x32_f16(
            false, aop[mt], false, bop, (short)0, acc[mt][nt], false, false);
    }
  }

  // epilogue
  const int rowbase = bm*128 + waveM*64;
  const int colbase = bn*128 + waveN*32;
  #pragma unroll
  for(int mt=0;mt<4;mt++)
    #pragma unroll
    for(int nt=0;nt<2;nt++)
      #pragma unroll
      for(int r=0;r<8;r++){
        int grow = rowbase + mt*16 + lh*8 + r;
        int gcol = colbase + nt*16 + l15;
        if (grow < M){
          float v = acc[mt][nt][r];
          if (bias) v += bias[gcol];
          if (act == 1) v = fmaxf(v, 0.f);
          else if (act == 2) v = 0.5f*v*(1.f + erff(v*0.70710678118654752f));
          if (res) v += res[(size_t)grow*N + gcol];
          out[(size_t)grow*N + gcol] = v;
        }
      }
}

// =====================================================================
// Embedding: xb[b,s,:] = tok_emb[id] + pos_emb[s]
// =====================================================================
__global__ __launch_bounds__(256) void embed_kernel(
    const int* __restrict__ ids, const float* __restrict__ tok,
    const float* __restrict__ pos, float* __restrict__ out)
{
  int i = blockIdx.x*256 + threadIdx.x;     // over MROWS*H_
  int e = i & (H_-1);
  int row = i >> 10;
  int s = row & (S_-1);
  out[i] = tok[(size_t)ids[row]*H_ + e] + pos[(size_t)s*H_ + e];
}

// =====================================================================
// Masked chunk average: cavg[b,c,e]
// =====================================================================
__global__ __launch_bounds__(256) void chunk_avg_kernel(
    const float* __restrict__ h, const int* __restrict__ ids, float* __restrict__ cavg)
{
  int i = blockIdx.x*256 + threadIdx.x;     // over 32*H_
  int e = i & (H_-1);
  int bc = i >> 10;                         // 0..31
  int b = bc >> 4, c = bc & 15;
  int base = b*S_ + c*CS_;
  float sum = 0.f, cnt = 0.f;
  for(int j=0;j<CS_;j++){
    float m = (ids[base+j] != 0) ? 1.f : 0.f;
    sum += h[(size_t)(base+j)*H_ + e] * m;
    cnt += m;
  }
  cavg[i] = sum / (cnt + 1e-10f);
}

// =====================================================================
// LayerNorm, one wave per row (H=1024, 32 elems/lane)
// =====================================================================
__global__ __launch_bounds__(128) void ln_kernel(
    const float* __restrict__ h, const float* __restrict__ g,
    const float* __restrict__ bta, float* __restrict__ out)
{
  int wave = threadIdx.x >> 5, lane = threadIdx.x & 31;
  int row = blockIdx.x*4 + wave;
  const float* x = h + (size_t)row*H_;
  float v[32];
  float s = 0.f;
  #pragma unroll
  for(int t=0;t<32;t++){ v[t] = x[lane + 32*t]; s += v[t]; }
  for(int m=16;m>=1;m>>=1) s += __shfl_xor(s, m, 32);
  float mean = s * (1.f/H_);
  float q = 0.f;
  #pragma unroll
  for(int t=0;t<32;t++){ float d = v[t]-mean; q += d*d; }
  for(int m=16;m>=1;m>>=1) q += __shfl_xor(q, m, 32);
  float rstd = rsqrtf(q*(1.f/H_) + 1e-5f);
  float* o = out + (size_t)row*H_;
  #pragma unroll
  for(int t=0;t<32;t++){
    int e = lane + 32*t;
    o[e] = (v[t]-mean)*rstd*g[e] + bta[e];
  }
}

// =====================================================================
// Chunk scores + top-8 selection -> bitmask per query
// =====================================================================
__global__ __launch_bounds__(128) void cscore_topk_kernel(
    const float* __restrict__ qe, const float* __restrict__ cenc,
    unsigned* __restrict__ cmask)
{
  int wave = threadIdx.x >> 5, lane = threadIdx.x & 31;
  int row = blockIdx.x*4 + wave;            // b*S + s
  int b = row >> 11;                        // / S_
  const float* q = qe + (size_t)row*H_;
  float qv[32];
  #pragma unroll
  for(int t=0;t<32;t++) qv[t] = q[lane + 32*t];
  float sc[NC_];
  for(int c=0;c<NC_;c++){
    const float* ce = cenc + (size_t)(b*NC_ + c)*H_;
    float s = 0.f;
    #pragma unroll
    for(int t=0;t<32;t++) s += qv[t]*ce[lane + 32*t];
    for(int m=16;m>=1;m>>=1) s += __shfl_xor(s, m, 32);
    sc[c] = s;
  }
  unsigned mask = 0;
  for(int k=0;k<TOPK_;k++){
    float best = -__builtin_inff(); int bi = 0;
    for(int c=0;c<NC_;c++)
      if(!((mask>>c)&1u) && sc[c] > best){ best = sc[c]; bi = c; }
    mask |= (1u << bi);
  }
  if (lane == 0) cmask[row] = mask;
}

// =====================================================================
// Sparse-chunk-masked flash attention.
// Block: 256 thr = 8 waves; wave = 16 queries x 64-key blocks (half chunk).
// QK^T and P.V via v_wmma_f32_16x16x32_f16; online softmax in f32.
// =====================================================================
#define KP 136   // Ks pitch (f16)
#define VP 72    // Vs / Ps pitch (f16)

__global__ __launch_bounds__(256) void attn_kernel(
    const float* __restrict__ q, const float* __restrict__ k, const float* __restrict__ v,
    const int* __restrict__ ids, const unsigned* __restrict__ cmask,
    float* __restrict__ ao)
{
  __shared__ _Float16 Ks[64*KP];    // [key][hd]
  __shared__ _Float16 Vs[128*VP];   // [hd][key] (transposed)
  __shared__ _Float16 Ps[128*VP];   // per-wave P staging [query][key]
  __shared__ float    kval[64];

  const int tid  = threadIdx.x;
  const int lane = tid & 31, wave = tid >> 5;
  const int l15  = lane & 15, lh = lane >> 4;
  const int bh   = blockIdx.y;            // b*NH + head
  const int b    = bh >> 3, head = bh & 7;
  const int qblk = blockIdx.x;            // 0..15
  const float scale = 0.08838834764831845f;  // HD^-0.5

  // Q A-operands straight from global (per-wave 16 rows, 4 K-steps of 32)
  v16h qa[4];
  {
    int qrow = qblk*128 + wave*16 + l15;
    const float4* qp4 = (const float4*)(q + (size_t)(b*S_ + qrow)*H_ + head*HD_);
    #pragma unroll
    for(int kk=0;kk<4;kk++){
      float4 f0 = qp4[(kk*32      + lh*8)>>2], f1 = qp4[((kk*32      + lh*8)>>2)+1];
      float4 f2 = qp4[(kk*32 + 16 + lh*8)>>2], f3 = qp4[((kk*32 + 16 + lh*8)>>2)+1];
      v8h lo = pack8(f0, f1), hi = pack8(f2, f3);
      #pragma unroll
      for(int j=0;j<8;j++){ qa[kk][j] = lo[j]; qa[kk][j+8] = hi[j]; }
    }
  }
  // top-k chunk masks for the 8 rows this lane covers in C-layout
  unsigned cm[8];
  #pragma unroll
  for(int r=0;r<8;r++){
    int qrow = qblk*128 + wave*16 + lh*8 + r;
    cm[r] = cmask[b*S_ + qrow];
  }

  float mrow[8], lrow[8];
  #pragma unroll
  for(int r=0;r<8;r++){ mrow[r] = -__builtin_inff(); lrow[r] = 0.f; }
  v8f oacc[8];
  #pragma unroll
  for(int ht=0;ht<8;ht++) oacc[ht] = zero8();

  // staging maps
  const int k_kr = tid >> 2;            // key 0..63 (K tile)
  const int k_c0 = (tid & 3) * 32;      // hd group
  const int v_kp = tid & 31;            // key pair (V tile): keys 2*v_kp, 2*v_kp+1
  const int v_c0 = (tid >> 5) * 16;     // hd 0..112

  for(int kb2=0;kb2<32;kb2++){
    int key0 = kb2*64;
    __syncthreads();
    // stage K tile [64][128], vectorized
    {
      const float4* kp4 = (const float4*)(k + (size_t)(b*S_ + key0 + k_kr)*H_ + head*HD_ + k_c0);
      v8h* dst = (v8h*)&Ks[k_kr*KP + k_c0];
      #pragma unroll
      for(int g=0; g<4; g++)
        dst[g] = pack8(kp4[2*g], kp4[2*g+1]);
    }
    // stage V tile transposed [hd][key], pack key pairs -> b32 stores
    {
      const float4* s0 = (const float4*)(v + (size_t)(b*S_ + key0 + 2*v_kp    )*H_ + head*HD_ + v_c0);
      const float4* s1 = (const float4*)(v + (size_t)(b*S_ + key0 + 2*v_kp + 1)*H_ + head*HD_ + v_c0);
      float4 a0 = s0[0], a1 = s0[1], a2 = s0[2], a3 = s0[3];
      float4 c0 = s1[0], c1 = s1[1], c2 = s1[2], c3 = s1[3];
      float w0[16] = {a0.x,a0.y,a0.z,a0.w,a1.x,a1.y,a1.z,a1.w,
                      a2.x,a2.y,a2.z,a2.w,a3.x,a3.y,a3.z,a3.w};
      float w1[16] = {c0.x,c0.y,c0.z,c0.w,c1.x,c1.y,c1.z,c1.w,
                      c2.x,c2.y,c2.z,c2.w,c3.x,c3.y,c3.z,c3.w};
      #pragma unroll
      for(int j=0;j<16;j++){
        v2h t; t[0] = (_Float16)w0[j]; t[1] = (_Float16)w1[j];
        *(v2h*)&Vs[(v_c0+j)*VP + 2*v_kp] = t;
      }
    }
    if (tid < 64) kval[tid] = (ids[b*S_ + key0 + tid] != 0) ? 1.f : 0.f;
    __syncthreads();

    // scores: 16 queries x 64 keys
    v8f sacc[4];
    #pragma unroll
    for(int nt=0;nt<4;nt++) sacc[nt] = zero8();
    #pragma unroll
    for(int kk=0;kk<4;kk++){
      #pragma unroll
      for(int nt=0;nt<4;nt++){
        const _Float16* p = &Ks[(nt*16 + l15)*KP + kk*32 + lh*8];
        v16h bop = mk_op(p, p+16);
        sacc[nt] = __builtin_amdgcn_wmma_f32_16x16x32_f16(
            false, qa[kk], false, bop, (short)0, sacc[nt], false, false);
      }
    }

    // mask + scale
    const int chunk = kb2 >> 1;
    float kvl[4];
    #pragma unroll
    for(int nt=0;nt<4;nt++) kvl[nt] = kval[nt*16 + l15];
    float pm[4][8];
    #pragma unroll
    for(int nt=0;nt<4;nt++)
      #pragma unroll
      for(int r=0;r<8;r++){
        bool sel = (((cm[r]>>chunk)&1u) != 0u) && (kvl[nt] > 0.f);
        pm[nt][r] = sel ? sacc[nt][r]*scale : -__builtin_inff();
      }

    // online softmax update (row stats within 16-lane halves)
    float corr[8];
    #pragma unroll
    for(int r=0;r<8;r++){
      float mx = fmaxf(fmaxf(pm[0][r],pm[1][r]), fmaxf(pm[2][r],pm[3][r]));
      for(int m=8;m>=1;m>>=1) mx = fmaxf(mx, __shfl_xor(mx, m, 32));
      float mn = fmaxf(mrow[r], mx);
      corr[r] = (mrow[r] == -__builtin_inff()) ? 0.f : __expf(mrow[r] - mn);
      float ls = 0.f;
      #pragma unroll
      for(int nt=0;nt<4;nt++){
        float p = (pm[nt][r] == -__builtin_inff()) ? 0.f : __expf(pm[nt][r] - mn);
        pm[nt][r] = p;
        ls += p;
      }
      for(int m=8;m>=1;m>>=1) ls += __shfl_xor(ls, m, 32);
      lrow[r] = lrow[r]*corr[r] + ls;
      mrow[r] = mn;
    }
    #pragma unroll
    for(int ht=0;ht<8;ht++)
      #pragma unroll
      for(int r=0;r<8;r++) oacc[ht][r] *= corr[r];

    // P: C-layout -> A-layout via per-wave LDS staging
    #pragma unroll
    for(int nt=0;nt<4;nt++)
      #pragma unroll
      for(int r=0;r<8;r++)
        Ps[(wave*16 + lh*8 + r)*VP + nt*16 + l15] = (_Float16)pm[nt][r];
    asm volatile("s_wait_dscnt 0" ::: "memory");

    // O += P.V
    #pragma unroll
    for(int kk2=0;kk2<2;kk2++){
      const _Float16* pp = &Ps[(wave*16 + l15)*VP + kk2*32 + lh*8];
      v16h pa = mk_op(pp, pp+16);
      #pragma unroll
      for(int ht=0;ht<8;ht++){
        const _Float16* vb = &Vs[(ht*16 + l15)*VP + kk2*32 + lh*8];
        v16h bop = mk_op(vb, vb+16);
        oacc[ht] = __builtin_amdgcn_wmma_f32_16x16x32_f16(
            false, pa, false, bop, (short)0, oacc[ht], false, false);
      }
    }
  }

  // normalize + store
  #pragma unroll
  for(int r=0;r<8;r++){
    float inv = 1.f/(lrow[r] + 1e-20f);
    int qrow = qblk*128 + wave*16 + lh*8 + r;
    float* op = ao + (size_t)(b*S_ + qrow)*H_ + head*HD_;
    #pragma unroll
    for(int ht=0;ht<8;ht++)
      op[ht*16 + l15] = oacc[ht][r]*inv;
  }
}

// =====================================================================
// Host side
// =====================================================================
static void gemm(hipStream_t s, const float* A, const float* W, const float* bias,
                 const float* res, float* out, int M, int N, int K, int act){
  dim3 g(N/128, (M+127)/128), blk(256);
  gemm_bias_act_kernel<<<g, blk, 0, s>>>(A, W, bias, res, out, M, N, K, act);
}

extern "C" void kernel_launch(void* const* d_in, const int* in_sizes, int n_in,
                              void* d_out, int out_size, void* d_ws, size_t ws_size,
                              hipStream_t stream)
{
  const int*   ids     = (const int*)  d_in[0];
  const float* tok_emb = (const float*)d_in[1];
  const float* pos_emb = (const float*)d_in[2];
  const float* in_w    = (const float*)d_in[3];
  const float* in_b    = (const float*)d_in[4];
  const float* ch_w1   = (const float*)d_in[5];
  const float* ch_b1   = (const float*)d_in[6];
  const float* ch_w2   = (const float*)d_in[7];
  const float* ch_b2   = (const float*)d_in[8];
  const float* qe_w1   = (const float*)d_in[9];
  const float* qe_b1   = (const float*)d_in[10];
  const float* qe_w2   = (const float*)d_in[11];
  const float* qe_b2   = (const float*)d_in[12];
  const float* q_w     = (const float*)d_in[13];
  const float* q_b     = (const float*)d_in[14];
  const float* k_w     = (const float*)d_in[15];
  const float* k_b     = (const float*)d_in[16];
  const float* v_w     = (const float*)d_in[17];
  const float* v_b     = (const float*)d_in[18];
  const float* o_w     = (const float*)d_in[19];
  const float* o_b     = (const float*)d_in[20];
  const float* f_w1    = (const float*)d_in[21];
  const float* f_b1    = (const float*)d_in[22];
  const float* f_w2    = (const float*)d_in[23];
  const float* f_b2    = (const float*)d_in[24];
  const float* ln1_g   = (const float*)d_in[25];
  const float* ln1_b   = (const float*)d_in[26];
  const float* ln2_g   = (const float*)d_in[27];
  const float* ln2_b   = (const float*)d_in[28];
  const float* out_w   = (const float*)d_in[29];
  const float* out_b   = (const float*)d_in[30];
  (void)in_sizes; (void)n_in; (void)out_size; (void)ws_size;

  float* ws = (float*)d_ws;
  size_t off = 0;
  float* h    = ws + off; off += (size_t)MROWS*H_;
  float* xb   = ws + off; off += (size_t)MROWS*H_;
  float* qe   = ws + off; off += (size_t)MROWS*H_;     // later reused as attention output
  float* mid  = ws + off; off += (size_t)MROWS*4*H_;   // FFN mid / qe mid
  float* qb   = ws + off; off += (size_t)MROWS*H_;
  float* kb   = ws + off; off += (size_t)MROWS*H_;
  float* vb   = ws + off; off += (size_t)MROWS*H_;
  float* cavg = ws + off; off += (size_t)32*H_;
  float* cmid = ws + off; off += (size_t)32*(H_/2);
  float* cenc = ws + off; off += (size_t)32*H_;
  unsigned* cmaskp = (unsigned*)(ws + off); off += MROWS;

  // embeddings + input projection
  embed_kernel<<<(MROWS*H_)/256, 256, 0, stream>>>(ids, tok_emb, pos_emb, xb);
  gemm(stream, xb, in_w, in_b, nullptr, h, MROWS, H_, H_, 0);

  // chunk encodings (once)
  chunk_avg_kernel<<<(32*H_)/256, 256, 0, stream>>>(h, ids, cavg);
  gemm(stream, cavg, ch_w1, ch_b1, nullptr, cmid, 32, H_/2, H_, 1);
  gemm(stream, cmid, ch_w2, ch_b2, nullptr, cenc, 32, H_, H_/2, 0);

  for(int i=0;i<L_;i++){
    // query encodings + top-k chunk selection
    gemm(stream, h,   qe_w1, qe_b1, nullptr, mid, MROWS, H_/2, H_, 1);
    gemm(stream, mid, qe_w2, qe_b2, nullptr, qe,  MROWS, H_, H_/2, 0);
    cscore_topk_kernel<<<MROWS/4, 128, 0, stream>>>(qe, cenc, cmaskp);

    // attention block
    ln_kernel<<<MROWS/4, 128, 0, stream>>>(h, ln1_g + (size_t)i*H_, ln1_b + (size_t)i*H_, xb);
    gemm(stream, xb, q_w + (size_t)i*H_*H_, q_b + (size_t)i*H_, nullptr, qb, MROWS, H_, H_, 0);
    gemm(stream, xb, k_w + (size_t)i*H_*H_, k_b + (size_t)i*H_, nullptr, kb, MROWS, H_, H_, 0);
    gemm(stream, xb, v_w + (size_t)i*H_*H_, v_b + (size_t)i*H_, nullptr, vb, MROWS, H_, H_, 0);
    attn_kernel<<<dim3(S_/128, B_*NH_), 256, 0, stream>>>(qb, kb, vb, ids, cmaskp, qe);
    gemm(stream, qe, o_w + (size_t)i*H_*H_, o_b + (size_t)i*H_, h, h, MROWS, H_, H_, 0);

    // FFN block
    ln_kernel<<<MROWS/4, 128, 0, stream>>>(h, ln2_g + (size_t)i*H_, ln2_b + (size_t)i*H_, xb);
    gemm(stream, xb,  f_w1 + (size_t)i*H_*4*H_, f_b1 + (size_t)i*4*H_, nullptr, mid, MROWS, 4*H_, H_, 2);
    gemm(stream, mid, f_w2 + (size_t)i*4*H_*H_, f_b2 + (size_t)i*H_,   h,       h,   MROWS, H_, 4*H_, 0);
  }

  // logits
  gemm(stream, h, out_w, out_b, nullptr, (float*)d_out, MROWS, V_, H_, 0);
}
